// AxialAttention_wopos_8143257993310
// MI455X (gfx1250) — matrix operand order repaired and checked by reading, not compiled, not run on hardware
//
#include <hip/hip_runtime.h>

// MI455X (gfx1250) AxialAttention-no-pos.
// Memory-bound problem (~15 GFLOP vs 128 MB HBM traffic -> ~5.5us floor at
// 23.3 TB/s). One block per (n,w) slice; 8 waves = 8 attention groups.
// GEMM1 + sv use v_wmma_f32_16x16x32_bf16; qk scores use v_wmma_f32_16x16x4_f32
// (full fp32 accuracy on the softmax input, K=8 inner dim).

typedef __attribute__((ext_vector_type(16))) __bf16 bf16x16;
typedef __attribute__((ext_vector_type(8)))  __bf16 bf16x8;
typedef __attribute__((ext_vector_type(8)))  float  f32x8;
typedef __attribute__((ext_vector_type(2)))  float  f32x2;

#define BN_EPS 1e-5f

// ---- LDS layout (dynamic, 229376 bytes total) ----
// [     0, 32768)  xT   : bf16 [h=128][c=128]   (WMMA-B layout for GEMM1)
// [ 32768, 98304)  wl   : bf16 [o=256][c=128]   (WMMA-A layout for GEMM1)
// [ 98304,131072)  qS   : f32  [g=8][i=128][c=8]
// [131072,163840)  kS   : f32  [g=8][j=128][c=8]
// [163840,196608)  vS   : bf16 [g=8][c=16][j=128]
// [196608,229376)  pS   : bf16 [g=8][i_loc=16][j=128]
#define SMEM_BYTES 229376

__device__ __forceinline__ bf16x16 ld_fragA(const __bf16* base) {
    // A operand in this code's contiguous k order (k = 16*lHi + s): per lane, 16 contiguous K at +0.
    bf16x8 lo = *(const bf16x8*)(base);
    bf16x8 hi = *(const bf16x8*)(base + 8);
    bf16x16 r;
#pragma unroll
    for (int i = 0; i < 8; ++i) { r[i] = lo[i]; r[i + 8] = hi[i]; }
    return r;
}

// Fold qkv-BN scale into the weight matrix (bf16) and produce the bias vector.
__global__ __launch_bounds__(256)
void fold_weights_kernel(const float* __restrict__ w,
                         const float* __restrict__ gamma,
                         const float* __restrict__ beta,
                         const float* __restrict__ mean,
                         const float* __restrict__ var,
                         __bf16* __restrict__ wbf,
                         float* __restrict__ bias) {
    int idx = blockIdx.x * 256 + threadIdx.x;     // 256*128 = 32768 elements
    if (idx >= 256 * 128) return;
    int o = idx >> 7;
    int c = idx & 127;
    float s = gamma[o] * rsqrtf(var[o] + BN_EPS);
    wbf[idx] = (__bf16)(w[idx] * s);
    if (c == 0) bias[o] = beta[o] - mean[o] * s;
}

__global__ __launch_bounds__(256)
void axial_attn_kernel(const float* __restrict__ x,
                       const __bf16* __restrict__ wbf,
                       const float* __restrict__ bias,
                       const float* __restrict__ sim_gamma,
                       const float* __restrict__ sim_beta,
                       const float* __restrict__ sim_mean,
                       const float* __restrict__ sim_var,
                       const float* __restrict__ out_gamma,
                       const float* __restrict__ out_beta,
                       const float* __restrict__ out_mean,
                       const float* __restrict__ out_var,
                       float* __restrict__ out) {
    extern __shared__ char smem[];
    __bf16* xT = (__bf16*)(smem);
    __bf16* wl = (__bf16*)(smem + 32768);
    float*  qS = (float*) (smem + 98304);
    float*  kS = (float*) (smem + 131072);
    __bf16* vS = (__bf16*)(smem + 163840);
    __bf16* pS = (__bf16*)(smem + 196608);

    const int b    = blockIdx.x;       // b = n*128 + w
    const int n    = b >> 7;
    const int w    = b & 127;
    const int tid  = threadIdx.x;
    const int g    = tid >> 5;         // wave id == attention group
    const int lane = tid & 31;
    const int lM   = lane & 15;
    const int lHi  = lane >> 4;
    const f32x8 fz = {};

    // ---------------- Phase 0: stage x slice (bf16) and folded weights ------
    for (int idx = tid; idx < 128 * 128; idx += 256) {
        int c = idx & 127;                         // lanes contiguous in c
        int h = idx >> 7;
        float v = x[((((size_t)n * 128 + c) * 128) + h) * 128 + w];  // L2-resident
        xT[h * 128 + c] = (__bf16)v;
    }
    {
        const uint4* src = (const uint4*)wbf;      // 64 KB folded weights
        uint4* dst = (uint4*)wl;
        for (int i = tid; i < 8192; i += 256) dst[i] = src[i];
    }
    __syncthreads();

    // ---------------- Phase 1: qkv = W' * x  (wave g -> o-tiles 2g, 2g+1) ---
    f32x8 acc[2][8];
#pragma unroll
    for (int t = 0; t < 2; ++t)
#pragma unroll
        for (int it = 0; it < 8; ++it) acc[t][it] = fz;

#pragma unroll
    for (int ks = 0; ks < 4; ++ks) {
        const int kbA = lHi * 16 + 32 * ks;
        bf16x16 a0 = ld_fragA(wl + (g * 32 +      lM) * 128 + kbA);
        bf16x16 a1 = ld_fragA(wl + (g * 32 + 16 + lM) * 128 + kbA);
#pragma unroll
        for (int it = 0; it < 8; ++it) {
            // B operand, same contiguous k order as A (k = 16*lHi + s): per lane, 16 contiguous K at lHi*16.
            bf16x16 bf = *(const bf16x16*)(xT + (it * 16 + lM) * 128 + lHi * 16 + 32 * ks);
            acc[0][it] = __builtin_amdgcn_wmma_f32_16x16x32_bf16(
                false, a0, false, bf, (short)0, acc[0][it], false, false);
            acc[1][it] = __builtin_amdgcn_wmma_f32_16x16x32_bf16(
                false, a1, false, bf, (short)0, acc[1][it], false, false);
        }
    }

    // bias (folded qkv-BN shift); D layout: row = r + 8*lHi, col = it*16+lM
    float bl[2][8];
#pragma unroll
    for (int t = 0; t < 2; ++t) {
        const float* bp = bias + g * 32 + t * 16 + 8 * lHi;
        float4 b0 = *(const float4*)(bp);
        float4 b1 = *(const float4*)(bp + 4);
        bl[t][0] = b0.x; bl[t][1] = b0.y; bl[t][2] = b0.z; bl[t][3] = b0.w;
        bl[t][4] = b1.x; bl[t][5] = b1.y; bl[t][6] = b1.z; bl[t][7] = b1.w;
    }

    // Scatter q (channels 0..7, lanes 0-15), k (8..15, lanes 16-31), v (16..31)
    float* qkDst = (lHi ? kS : qS) + g * 1024;     // [i or j][c=8]
    __bf16* vg_w = vS + g * 2048;                   // [c=16][j=128]
#pragma unroll
    for (int it = 0; it < 8; ++it) {
        const int i = it * 16 + lM;
#pragma unroll
        for (int r = 0; r < 8; ++r) {
            qkDst[i * 8 + r] = acc[0][it][r] + bl[0][r];
            vg_w[(8 * lHi + r) * 128 + i] = (__bf16)(acc[1][it][r] + bl[1][r]);
        }
    }
    // No barrier: wave g produced and is the sole consumer of group g data;
    // LDS ops from the same wave complete in order.

    // ---------------- Phase 2: attention for group g ------------------------
    const float a_s = sim_gamma[g] * rsqrtf(sim_var[g] + BN_EPS);
    const float b_s = sim_beta[g] - sim_mean[g] * a_s;

    float ao[8], bo[8];
    {
        const int oc = g * 16 + 8 * lHi;
        float4 gg0 = *(const float4*)(out_gamma + oc), gg1 = *(const float4*)(out_gamma + oc + 4);
        float4 bb0 = *(const float4*)(out_beta  + oc), bb1 = *(const float4*)(out_beta  + oc + 4);
        float4 mm0 = *(const float4*)(out_mean  + oc), mm1 = *(const float4*)(out_mean  + oc + 4);
        float4 vv0 = *(const float4*)(out_var   + oc), vv1 = *(const float4*)(out_var   + oc + 4);
        float gga[8] = {gg0.x,gg0.y,gg0.z,gg0.w,gg1.x,gg1.y,gg1.z,gg1.w};
        float bba[8] = {bb0.x,bb0.y,bb0.z,bb0.w,bb1.x,bb1.y,bb1.z,bb1.w};
        float mma[8] = {mm0.x,mm0.y,mm0.z,mm0.w,mm1.x,mm1.y,mm1.z,mm1.w};
        float vva[8] = {vv0.x,vv0.y,vv0.z,vv0.w,vv1.x,vv1.y,vv1.z,vv1.w};
#pragma unroll
        for (int r = 0; r < 8; ++r) {
            float s = gga[r] * rsqrtf(vva[r] + BN_EPS);
            ao[r] = s;
            bo[r] = bba[r] - mma[r] * s;
        }
    }

    const float*  qg = qS + g * 1024;
    const float*  kg = kS + g * 1024;
    const __bf16* vg = vS + g * 2048;
    __bf16*       pg = pS + g * 2048;

    for (int it = 0; it < 8; ++it) {
        // ---- S row block (16 x 128): fp32 WMMA, K=8 over q/k channels ----
        f32x8 s[8];
#pragma unroll
        for (int jt = 0; jt < 8; ++jt) s[jt] = fz;
#pragma unroll
        for (int kk = 0; kk < 2; ++kk) {
            // f32 A 16x4: lane holds K = lHi*2, lHi*2+1 within 4-wide chunk
            f32x2 aq = *(const f32x2*)(qg + (it * 16 + lM) * 8 + 4 * kk + lHi * 2);
#pragma unroll
            for (int jt = 0; jt < 8; ++jt) {
                f32x2 bk = *(const f32x2*)(kg + (jt * 16 + lM) * 8 + 4 * kk + lHi * 2);
                s[jt] = __builtin_amdgcn_wmma_f32_16x16x4_f32(
                    false, aq, false, bk, (short)0, s[jt], false, false);
            }
        }
        // sim BN affine (per-group scalars)
#pragma unroll
        for (int jt = 0; jt < 8; ++jt)
#pragma unroll
            for (int r = 0; r < 8; ++r) s[jt][r] = fmaf(a_s, s[jt][r], b_s);

        // ---- exact row softmax (row = r + 8*lHi; 16 lanes hold 8 j each) --
#pragma unroll
        for (int r = 0; r < 8; ++r) {
            float m = s[0][r];
#pragma unroll
            for (int jt = 1; jt < 8; ++jt) m = fmaxf(m, s[jt][r]);
#pragma unroll
            for (int msk = 1; msk < 16; msk <<= 1)
                m = fmaxf(m, __shfl_xor(m, msk, 32));   // stays in 16-lane half
            float sum = 0.f;
#pragma unroll
            for (int jt = 0; jt < 8; ++jt) {
                float e = __expf(s[jt][r] - m);
                s[jt][r] = e;
                sum += e;
            }
#pragma unroll
            for (int msk = 1; msk < 16; msk <<= 1)
                sum += __shfl_xor(sum, msk, 32);
            float inv = 1.f / sum;
            const int li = r + 8 * lHi;
#pragma unroll
            for (int jt = 0; jt < 8; ++jt)
                pg[li * 128 + jt * 16 + lM] = (__bf16)(s[jt][r] * inv);
        }

        // ---- sv tile: (16c x 16i) = v(16x128) * P^T(128x16), bf16 WMMA ----
        f32x8 o = fz;
#pragma unroll
        for (int ks = 0; ks < 4; ++ks) {
            bf16x16 va = ld_fragA(vg + lM * 128 + lHi * 16 + 32 * ks);
            bf16x16 pb = *(const bf16x16*)(pg + lM * 128 + lHi * 16 + 32 * ks);
            o = __builtin_amdgcn_wmma_f32_16x16x32_bf16(
                false, va, false, pb, (short)0, o, false, false);
        }

        // ---- out BN + transposed store to (N, O, H, W) ----
#pragma unroll
        for (int r = 0; r < 8; ++r) {
            const int oc = g * 16 + 8 * lHi + r;
            const int i  = it * 16 + lM;
            out[((((size_t)n * 128 + oc) * 128) + i) * 128 + w] =
                fmaf(ao[r], o[r], bo[r]);
        }
    }
}

extern "C" void kernel_launch(void* const* d_in, const int* in_sizes, int n_in,
                              void* d_out, int out_size, void* d_ws, size_t ws_size,
                              hipStream_t stream) {
    const float* x         = (const float*)d_in[0];
    const float* w_qkv     = (const float*)d_in[1];
    const float* qkv_gamma = (const float*)d_in[2];
    const float* qkv_beta  = (const float*)d_in[3];
    const float* qkv_mean  = (const float*)d_in[4];
    const float* qkv_var   = (const float*)d_in[5];
    const float* sim_gamma = (const float*)d_in[6];
    const float* sim_beta  = (const float*)d_in[7];
    const float* sim_mean  = (const float*)d_in[8];
    const float* sim_var   = (const float*)d_in[9];
    const float* out_gamma = (const float*)d_in[10];
    const float* out_beta  = (const float*)d_in[11];
    const float* out_mean  = (const float*)d_in[12];
    const float* out_var   = (const float*)d_in[13];
    float* out = (float*)d_out;

    // workspace: [0,64K) bf16 folded weights, [64K,65K) f32 bias
    __bf16* wbf  = (__bf16*)d_ws;
    float*  bias = (float*)((char*)d_ws + 65536);

    fold_weights_kernel<<<128, 256, 0, stream>>>(
        w_qkv, qkv_gamma, qkv_beta, qkv_mean, qkv_var, wbf, bias);

    axial_attn_kernel<<<1024, 256, SMEM_BYTES, stream>>>(
        x, wbf, bias,
        sim_gamma, sim_beta, sim_mean, sim_var,
        out_gamma, out_beta, out_mean, out_var,
        out);
}
